// Attention2_57054345560466
// MI455X (gfx1250) — matrix operand order repaired and checked by reading, not compiled, not run on hardware
//
#include <hip/hip_runtime.h>
#include <hip/hip_bf16.h>
#include <stdint.h>

// ---------------------------------------------------------------------------
// CDNA5 (gfx1250) attention block.
//  - all matrix math via v_wmma_f32_16x16x32_bf16 (wave32)
//  - all LDS tiles laid out k-contiguous per fragment lane -> ds_load_b128
//  - tile staging via the Tensor Data Mover (tensor_load_to_lds), double
//    buffered, synchronized with s_wait_tensorcnt
// ---------------------------------------------------------------------------

typedef __bf16 bf16_t;
typedef __attribute__((ext_vector_type(16))) __bf16 v16bf;
typedef __attribute__((ext_vector_type(2)))  __bf16 v2bf;
typedef __attribute__((ext_vector_type(8)))  float  v8f;
typedef __attribute__((ext_vector_type(4)))  unsigned int v4u;
typedef __attribute__((ext_vector_type(8)))  int v8i;
typedef __attribute__((ext_vector_type(4)))  int v4i;

union FragBF {           // one 16x32 A-frag or 32x16 B-frag (8 VGPRs of bf16x2)
  v16bf    v;
  bf16_t   e[16];
  uint32_t u[8];
};

// Native f32 -> bf16 conversion (clang fptrunc; hardware bf16 cvt on gfx1250)
static __device__ __forceinline__ bf16_t f2bf(float f) { return (bf16_t)f; }

// Pack two f32 -> bf16x2 dword via a native vector convert; lets the backend
// pick the packed bf16 convert instead of scalar mask/or sequences.
static __device__ __forceinline__ uint32_t pack2bf(float lo, float hi) {
  v2bf p;
  p[0] = (bf16_t)lo;
  p[1] = (bf16_t)hi;
  uint32_t u;
  __builtin_memcpy(&u, &p, 4);
  return u;
}

// A-matrix (16x32, 16-bit) pair-base k for VGPR r, lane-half g (ISA 7.12.2)
static __device__ __forceinline__ int a_k(int r, int g) {
  return ((r < 4) ? (2 * r) : (16 + 2 * (r - 4))) + 8 * g;
}
// B-matrix (32x16, 16-bit) pair-base k for VGPR r, lane-half g (same order as A)
static __device__ __forceinline__ int b_k(int r, int g) { return a_k(r, g); }

// ---------------------------------------------------------------------------
// Tensor Data Mover: 2D tile load (tile == described tensor window).
// D# packing per ISA ch.8: group0 {count=1, lds_addr, global_addr, type=2},
// group1 {data_size=2B, pad_enable, pad_interval/amount, dims, tiles, stride}.
// This toolchain exposes the 6-arg builtin (extra int32x8 group before cpol).
// ---------------------------------------------------------------------------
static __device__ __forceinline__ void tdm_load_2d(
    uint32_t lds_off, const void* gaddr, uint32_t dim0, uint32_t dim1,
    uint32_t stride0, uint32_t pad_interval, uint32_t pad_amount) {
  uint64_t ga = (uint64_t)(uintptr_t)gaddr;
  v4u g0;
  g0[0] = 1u;                                   // count=1, gather off
  g0[1] = lds_off;                              // LDS byte address
  g0[2] = (uint32_t)ga;                         // global_addr[31:0]
  g0[3] = (uint32_t)((ga >> 32) & 0x01FFFFFFu) | (2u << 30);  // addr hi | type=2
  v8i g1;
  g1[0] = (int)((1u << 16)                      // data_size = 2 bytes
                | (1u << 20)                    // pad_enable
                | (pad_interval << 22) | (pad_amount << 25));
  g1[1] = (int)(dim0 << 16);                            // tensor_dim0 lo
  g1[2] = (int)((dim0 >> 16) | (dim1 << 16));           // dim0 hi | dim1 lo
  g1[3] = (int)((dim1 >> 16) | ((dim0 & 0xFFFFu) << 16));  // dim1 hi | tile_dim0
  g1[4] = (int)(dim1 & 0xFFFFu);                        // tile_dim1, tile_dim2=0
  g1[5] = (int)stride0;                                 // tensor_dim0_stride lo
  g1[6] = 0;
  g1[7] = 0;
  v4i z4 = {0, 0, 0, 0};
  v8i z8 = {0, 0, 0, 0, 0, 0, 0, 0};
  __builtin_amdgcn_tensor_load_to_lds(g0, g1, z4, z4, z8, 0);
}
static __device__ __forceinline__ uint32_t lds_off(const void* p) {
  return (uint32_t)(uintptr_t)p;   // generic LDS addr: low 32 bits = LDS offset
}

// ---------------------------------------------------------------------------
// fp32 -> bf16 conversion, 4 elements/thread, packed b64 stores
// ---------------------------------------------------------------------------
__global__ void convert_f32_bf16(const float* __restrict__ in,
                                 uint32_t* __restrict__ out, int n4) {
  int i = blockIdx.x * blockDim.x + threadIdx.x;
  int stride = gridDim.x * blockDim.x;
  for (; i < n4; i += stride) {
    float4 f = ((const float4*)in)[i];
    uint2 p;
    p.x = pack2bf(f.x, f.y);
    p.y = pack2bf(f.z, f.w);
    ((uint2*)out)[i] = p;
  }
}

// fp32 [R][Cc] -> bf16 [Cc][R] transpose (32x32 LDS tiles, 256 threads)
__global__ __launch_bounds__(256) void transpose_f32_bf16(
    const float* __restrict__ in, bf16_t* __restrict__ out, int R, int Cc) {
  __shared__ float tile[32][33];
  int bx = blockIdx.x * 32, by = blockIdx.y * 32;
  int tx = threadIdx.x & 31, ty = threadIdx.x >> 5;
  for (int i = ty; i < 32; i += 8)
    tile[i][tx] = in[(size_t)(by + i) * Cc + bx + tx];
  __syncthreads();
  for (int i = ty; i < 32; i += 8)
    out[(size_t)(bx + i) * R + by + tx] = f2bf(tile[tx][i]);
}

// ---------------------------------------------------------------------------
// Tiled bf16 GEMM: C[M,N] = A[M,K] * BT[N,K]^T  (+bias if F32OUT)
// Block tile 128x128, 8 waves (4 M x 2 N), wave tile 32x64, K-step 32.
// A and B tiles are both [row][k] in LDS (stride 40 = 80B, 16B multiple),
// staged by the TDM with double buffering.
// ---------------------------------------------------------------------------
template <bool F32OUT>
__global__ __launch_bounds__(256) void gemm_bf16_wmma(
    const bf16_t* __restrict__ A, const bf16_t* __restrict__ BT,
    void* __restrict__ Cout, const float* __restrict__ bias,
    int M, int N, int K) {
  __shared__ __align__(16) bf16_t As[2][128][40];
  __shared__ __align__(16) bf16_t Bs[2][128][40];

  const int tid  = threadIdx.x;
  const int lane = tid & 31;
  const int w    = tid >> 5;
  const int wm   = w & 3;
  const int wn   = w >> 2;
  const int nl   = lane & 15;
  const int g    = lane >> 4;

  const int m0 = blockIdx.y * 128;
  const int n0 = blockIdx.x * 128;
  const int KT = K >> 5;

  v8f acc[2][4] = {};

  // Prologue: TDM tiles for k0 = 0. 32-elem rows (16 dwords) -> pad_interval=3
  // (16 dwords), pad_amount=3 (4 dwords) => LDS row stride 40 elements.
  if (w == 0)
    tdm_load_2d(lds_off(&As[0][0][0]), A + (size_t)m0 * K, 32, 128, K, 3, 3);
  else if (w == 1)
    tdm_load_2d(lds_off(&Bs[0][0][0]), BT + (size_t)n0 * K, 32, 128, K, 3, 3);

  for (int it = 0; it < KT; ++it) {
    const int buf = it & 1;
    if (it + 1 < KT) {
      const int k1 = (it + 1) << 5;
      if (w == 0)
        tdm_load_2d(lds_off(&As[buf ^ 1][0][0]), A + (size_t)m0 * K + k1,
                    32, 128, K, 3, 3);
      else if (w == 1)
        tdm_load_2d(lds_off(&Bs[buf ^ 1][0][0]), BT + (size_t)n0 * K + k1,
                    32, 128, K, 3, 3);
      if (w < 2) __builtin_amdgcn_s_wait_tensorcnt(1);
    } else if (w < 2) {
      __builtin_amdgcn_s_wait_tensorcnt(0);
    }
    __syncthreads();   // tile `it` visible to all waves

    FragBF af[2];
    for (int sm = 0; sm < 2; sm++)
      for (int r = 0; r < 8; r++)
        af[sm].u[r] =
            *(const uint32_t*)&As[buf][wm * 32 + sm * 16 + nl][a_k(r, g)];
    for (int sn = 0; sn < 4; sn++) {
      FragBF bfr;
      for (int r = 0; r < 8; r++)
        bfr.u[r] =
            *(const uint32_t*)&Bs[buf][wn * 64 + sn * 16 + nl][b_k(r, g)];
      for (int sm = 0; sm < 2; sm++)
        acc[sm][sn] = __builtin_amdgcn_wmma_f32_16x16x32_bf16(
            false, af[sm].v, false, bfr.v, (short)0, acc[sm][sn], false, false);
    }
    __syncthreads();   // all reads of buf done before TDM refills it
  }

  for (int sm = 0; sm < 2; sm++)
    for (int sn = 0; sn < 4; sn++)
      for (int r = 0; r < 8; r++) {
        int grow = m0 + wm * 32 + sm * 16 + r + 8 * g;
        int gcol = n0 + wn * 64 + sn * 16 + nl;
        float v = acc[sm][sn][r];
        if (F32OUT)
          ((float*)Cout)[(size_t)grow * N + gcol] = v + bias[gcol];
        else
          ((bf16_t*)Cout)[(size_t)grow * N + gcol] = f2bf(v);
      }
}

// ---------------------------------------------------------------------------
// Flash attention. Computes S^T = K * Q^T per tile so every WMMA operand is a
// k-contiguous LDS read, and the softmax axis (j) lands on the VGPR dimension:
// row reduce = VALU max3 tree + one shfl_xor(16).
// Grid: (N/128, H). 8 waves; wave w owns query rows [ibase, ibase+16).
// ---------------------------------------------------------------------------
__global__ __launch_bounds__(256) void flash_attn_bf16(
    const bf16_t* __restrict__ QK,   // [4096][2048]  Q cols [0,1024), K cols [1024,2048)
    const bf16_t* __restrict__ Vt,   // [1024][4096]  Vt[h*64+d][n]
    bf16_t* __restrict__ O) {        // [4096][1024]
  const int   N = 4096, CQK = 2048, C = 1024;
  const float SCALE = 0.125f;        // 64^-0.5

  __shared__ __align__(16) bf16_t Ks[2][32][72];   // K tile  [j][d], 144B rows
  __shared__ __align__(16) bf16_t Vs[2][64][40];   // Vt tile [d][j],  80B rows
  __shared__ __align__(16) bf16_t Ps[8][16][40];   // P scratch [i][j] per wave

  const int tid  = threadIdx.x;
  const int lane = tid & 31;
  const int w    = tid >> 5;
  const int nl   = lane & 15;
  const int g    = lane >> 4;

  const int h     = blockIdx.y;
  const int hbase = h * 64;
  const int ibase = blockIdx.x * 128 + w * 16;

  // Q as B-fragments (B = Q^T, 32d x 16i): lane holds column i = ibase + nl
  FragBF qb[2];
  {
    const bf16_t* Qrow = QK + (size_t)(ibase + nl) * CQK + hbase;
    for (int t = 0; t < 2; t++)
      for (int r = 0; r < 8; r++)
        qb[t].u[r] = *(const uint32_t*)(Qrow + 32 * t + b_k(r, g));
  }

  float m_run = -1e30f, l_run = 0.0f;  // softmax state for column i = ibase+nl
  v8f Oacc[4] = {};                    // O tile 16i x 64d (rows m = i)

  // Prologue TDM: K rows j=0..31 (64-elem rows -> pad_interval=4, amount=3,
  // LDS stride 72), Vt rows d (32-elem rows -> stride 40).
  if (w == 0)
    tdm_load_2d(lds_off(&Ks[0][0][0]),
                QK + (size_t)0 * CQK + 1024 + hbase, 64, 32, CQK, 4, 3);
  else if (w == 1)
    tdm_load_2d(lds_off(&Vs[0][0][0]),
                Vt + (size_t)hbase * N, 32, 64, N, 3, 3);

  for (int it = 0; it < N / 32; ++it) {
    const int buf = it & 1;
    if (it + 1 < N / 32) {
      const int jn = (it + 1) * 32;
      if (w == 0)
        tdm_load_2d(lds_off(&Ks[buf ^ 1][0][0]),
                    QK + (size_t)jn * CQK + 1024 + hbase, 64, 32, CQK, 4, 3);
      else if (w == 1)
        tdm_load_2d(lds_off(&Vs[buf ^ 1][0][0]),
                    Vt + (size_t)hbase * N + jn, 32, 64, N, 3, 3);
      if (w < 2) __builtin_amdgcn_s_wait_tensorcnt(1);
    } else if (w < 2) {
      __builtin_amdgcn_s_wait_tensorcnt(0);
    }
    __syncthreads();

    // S^T tiles: ST0 rows j=0..15, ST1 rows j=16..31 (cols i)
    v8f ST0 = {}, ST1 = {};
    for (int kt = 0; kt < 2; kt++) {
      FragBF ka0, ka1;   // A = K tile rows j, k = d
      for (int r = 0; r < 8; r++) {
        ka0.u[r] = *(const uint32_t*)&Ks[buf][nl][32 * kt + a_k(r, g)];
        ka1.u[r] = *(const uint32_t*)&Ks[buf][16 + nl][32 * kt + a_k(r, g)];
      }
      ST0 = __builtin_amdgcn_wmma_f32_16x16x32_bf16(
          false, ka0.v, false, qb[kt].v, (short)0, ST0, false, false);
      ST1 = __builtin_amdgcn_wmma_f32_16x16x32_bf16(
          false, ka1.v, false, qb[kt].v, (short)0, ST1, false, false);
    }

    // Online softmax over j (m-dimension of S^T): local 16-value tree + one
    // cross-half shuffle. Lane's 16 j's: {8g..8g+7} U {16+8g..16+8g+7}.
    float mx = -1e30f;
    for (int r = 0; r < 8; r++)
      mx = fmaxf(mx, fmaxf(ST0[r] * SCALE, ST1[r] * SCALE));
    mx = fmaxf(mx, __shfl_xor(mx, 16, 32));
    float mn    = fmaxf(m_run, mx);
    float alpha = __expf(m_run - mn);
    m_run = mn;

    float rs = 0.0f;
    uint32_t pk0[4], pk1[4];
    for (int r = 0; r < 4; r++) {
      float a0 = __expf(ST0[2 * r] * SCALE - mn);
      float a1 = __expf(ST0[2 * r + 1] * SCALE - mn);
      float b0 = __expf(ST1[2 * r] * SCALE - mn);
      float b1 = __expf(ST1[2 * r + 1] * SCALE - mn);
      rs += (a0 + a1) + (b0 + b1);
      pk0[r] = pack2bf(a0, a1);
      pk1[r] = pack2bf(b0, b1);
    }
    rs += __shfl_xor(rs, 16, 32);
    l_run = l_run * alpha + rs;

    // Store P[i][j]: lane's j-range is contiguous -> one b128 store per tile
    *(uint4*)&Ps[w][nl][8 * g]      = *(uint4*)pk0;   // j in [8g, 8g+8)
    *(uint4*)&Ps[w][nl][16 + 8 * g] = *(uint4*)pk1;   // j in [16+8g, 16+8g+8)

    // Rescale O: O rows m = r + 8g need alpha of lane (r + 8g)
    float arow[8];
    for (int r = 0; r < 8; r++) arow[r] = __shfl(alpha, r + 8 * g, 32);
    for (int t = 0; t < 4; t++)
      for (int r = 0; r < 8; r++) Oacc[t][r] *= arow[r];

    // P A-frag (intra-wave LDS store->load, in order, no barrier needed)
    FragBF pf;
    for (int r = 0; r < 8; r++)
      pf.u[r] = *(const uint32_t*)&Ps[w][nl][a_k(r, g)];

    // O(16i x 64d) += P(16i x 32j) * V(32j x 64d); B from Vt tile [d][j]
    for (int t = 0; t < 4; t++) {
      FragBF bv;
      for (int r = 0; r < 8; r++)
        bv.u[r] = *(const uint32_t*)&Vs[buf][16 * t + nl][b_k(r, g)];
      Oacc[t] = __builtin_amdgcn_wmma_f32_16x16x32_bf16(
          false, pf.v, false, bv.v, (short)0, Oacc[t], false, false);
    }
    __syncthreads();
  }

  // Normalize and store (bf16, consumed by the projection GEMM)
  float lrow[8];
  for (int r = 0; r < 8; r++) lrow[r] = __shfl(l_run, r + 8 * g, 32);
  for (int r = 0; r < 8; r++) {
    float inv = 1.0f / lrow[r];
    int grow = ibase + r + 8 * g;
    for (int t = 0; t < 4; t++)
      O[(size_t)grow * C + hbase + 16 * t + nl] = f2bf(Oacc[t][r] * inv);
  }
}

// ---------------------------------------------------------------------------
// Launch
// ---------------------------------------------------------------------------
extern "C" void kernel_launch(void* const* d_in, const int* in_sizes, int n_in,
                              void* d_out, int out_size, void* d_ws,
                              size_t ws_size, hipStream_t stream) {
  (void)in_sizes; (void)n_in; (void)out_size; (void)ws_size;
  const float* x1    = (const float*)d_in[0];   // [1,4096,1024]
  const float* x2    = (const float*)d_in[1];   // [1,4096,1024]
  const float* Wqkv  = (const float*)d_in[2];   // [1024,2048]
  const float* Wproj = (const float*)d_in[3];   // [1024,1024]
  const float* bproj = (const float*)d_in[4];   // [1024]

  char* ws = (char*)d_ws;
  bf16_t* X1b    = (bf16_t*)(ws);                          // 8 MB  [4096][1024]
  bf16_t* VtB    = (bf16_t*)(ws + (size_t)(8u  << 20));    // 8 MB  [1024][4096]
  bf16_t* WqkvT  = (bf16_t*)(ws + (size_t)(16u << 20));    // 4 MB  [2048][1024]
  bf16_t* WprojT = (bf16_t*)(ws + (size_t)(20u << 20));    // 2 MB  [1024][1024]
  bf16_t* QKb    = (bf16_t*)(ws + (size_t)(22u << 20));    // 16 MB [4096][2048]
  bf16_t* Ob     = (bf16_t*)(ws + (size_t)(38u << 20));    // 8 MB  [4096][1024]

  convert_f32_bf16<<<512, 256, 0, stream>>>(x1, (uint32_t*)X1b,
                                            4096 * 1024 / 4);
  transpose_f32_bf16<<<dim3(1024 / 32, 4096 / 32), 256, 0, stream>>>(
      x2, VtB, 4096, 1024);
  transpose_f32_bf16<<<dim3(2048 / 32, 1024 / 32), 256, 0, stream>>>(
      Wqkv, WqkvT, 1024, 2048);
  transpose_f32_bf16<<<dim3(1024 / 32, 1024 / 32), 256, 0, stream>>>(
      Wproj, WprojT, 1024, 1024);

  // QK projection: [4096,1024] x [1024,2048] -> bf16 [4096,2048]
  gemm_bf16_wmma<false><<<dim3(2048 / 128, 4096 / 128), 256, 0, stream>>>(
      X1b, WqkvT, (void*)QKb, nullptr, 4096, 2048, 1024);

  // Flash attention over 16 heads
  flash_attn_bf16<<<dim3(4096 / 128, 16), 256, 0, stream>>>(QKb, VtB, Ob);

  // Output projection + bias: [4096,1024] x [1024,1024] -> f32 d_out
  gemm_bf16_wmma<true><<<dim3(1024 / 128, 4096 / 128), 256, 0, stream>>>(
      Ob, WprojT, d_out, bproj, 4096, 1024, 1024);
}
